// DenseBlock_47914655154400
// MI455X (gfx1250) — matrix operand order repaired and checked by reading, compiled, not run on hardware
//
#include <hip/hip_runtime.h>

#define LAYERS 6
#define C_IN   128
#define GROW   32
#define HH     56
#define WW     56
#define NIMG   32
#define C_OUT  320           // 128 + 6*32
#define BN_EPS 1e-3f

typedef __bf16 bf16_t;
typedef bf16_t v8bf  __attribute__((ext_vector_type(8)));
typedef bf16_t v16bf __attribute__((ext_vector_type(16)));
typedef float  v8f   __attribute__((ext_vector_type(8)));

// ---------------------------------------------------------------------------
// Prep: fold BN into scale/shift:  y = x*a + b ;  a = g*rsqrt(v+eps), b = be - mu*a
// ---------------------------------------------------------------------------
__global__ void bn_prep_kernel(const float* __restrict__ g, const float* __restrict__ be,
                               const float* __restrict__ mu, const float* __restrict__ va,
                               float* __restrict__ a, float* __restrict__ b, int c) {
    int i = blockIdx.x * blockDim.x + threadIdx.x;
    if (i < c) {
        float s = g[i] * rsqrtf(va[i] + BN_EPS);
        a[i] = s;
        b[i] = be[i] - mu[i] * s;
    }
}

// ---------------------------------------------------------------------------
// Prep: repack HWIO fp32 (3,3,c,32) weights -> bf16 WMMA-B order.
// dst[((t*nq+q)*32 + o)*32 + k] = w[(t*c + q*32+k)*32 + o]
// (column-major per fragment: one column's 32 K-values are contiguous)
// ---------------------------------------------------------------------------
__global__ void pack_w_kernel(const float* __restrict__ wsrc, bf16_t* __restrict__ wdst, int c) {
    int nq = c >> 5;
    int total = 9 * nq * 32 * 32;
    for (int idx = blockIdx.x * blockDim.x + threadIdx.x; idx < total;
         idx += gridDim.x * blockDim.x) {
        int k  = idx & 31;
        int o  = (idx >> 5) & 31;
        int tq = idx >> 10;            // t*nq + q
        int q  = tq % nq;
        int t  = tq / nq;
        int ci = (q << 5) + k;
        wdst[idx] = (bf16_t)wsrc[(t * c + ci) * 32 + o];
    }
}

// ---------------------------------------------------------------------------
// Copy x (NHWC,128) into the concat buffer (NHWC,320), channels [0,128)
// ---------------------------------------------------------------------------
__global__ void copy_x_kernel(const float4* __restrict__ x, float* __restrict__ out) {
    const long total = (long)NIMG * HH * WW * (C_IN / 4);
    for (long u = blockIdx.x * (long)blockDim.x + threadIdx.x; u < total;
         u += (long)gridDim.x * blockDim.x) {
        long pos = u >> 5;             // 32 float4 per position
        int  c4  = (int)(u & 31);
        float4 v = x[u];
        *(float4*)(out + pos * C_OUT + c4 * 4) = v;
    }
}

// ---------------------------------------------------------------------------
// One dense layer: BN -> ReLU -> 3x3 SAME conv (c -> 32), fused, implicit GEMM.
// Block = 64 threads (2 waves), one 8x8 spatial tile of one image.
// Each wave register-blocks 2 M-tiles x 2 N-tiles (4 accumulators), so every
// B fragment is reused twice and each K-step issues 4 WMMAs.
// ---------------------------------------------------------------------------
__global__ __launch_bounds__(64) void dense_layer_kernel(
    const float* __restrict__ io,      // concat buffer, reads channels [0,c)
    float*       __restrict__ outb,    // same buffer, writes channels [c,c+32)
    const bf16_t* __restrict__ wpk,    // packed bf16 weights (9 * nq * 1024)
    const float* __restrict__ bnA, const float* __restrict__ bnB,
    const float* __restrict__ bias, int c) {
    extern __shared__ __align__(16) char smem[];

    const int nq      = c >> 5;
    const int strideB = c * 2 + 16;            // bytes per patch position (pad vs banks)
    const int tid     = threadIdx.x;
    const int n       = blockIdx.z;
    const int h0      = blockIdx.y * 8;
    const int w0      = blockIdx.x * 8;

    // ---- stage relu(bn(x)) for the 10x10 halo patch into LDS as bf16 ----
    const int cu    = c >> 3;                  // 8-channel units per position
    const int total = 100 * cu;
    for (int u = tid; u < total; u += 64) {
        int pos = u / cu;
        int ch  = (u - pos * cu) << 3;
        int ph = pos / 10, pw = pos - ph * 10;
        int gh = h0 + ph - 1, gw = w0 + pw - 1;
        v8bf rv = {};
        if ((unsigned)gh < HH && (unsigned)gw < WW) {
            const float* p = io + (((long)n * HH + gh) * WW + gw) * C_OUT + ch;
            float4 f0 = *(const float4*)(p);
            float4 f1 = *(const float4*)(p + 4);
            const float* A = bnA + ch;
            const float* B = bnB + ch;
            rv[0] = (bf16_t)fmaxf(fmaf(f0.x, A[0], B[0]), 0.0f);
            rv[1] = (bf16_t)fmaxf(fmaf(f0.y, A[1], B[1]), 0.0f);
            rv[2] = (bf16_t)fmaxf(fmaf(f0.z, A[2], B[2]), 0.0f);
            rv[3] = (bf16_t)fmaxf(fmaf(f0.w, A[3], B[3]), 0.0f);
            rv[4] = (bf16_t)fmaxf(fmaf(f1.x, A[4], B[4]), 0.0f);
            rv[5] = (bf16_t)fmaxf(fmaf(f1.y, A[5], B[5]), 0.0f);
            rv[6] = (bf16_t)fmaxf(fmaf(f1.z, A[6], B[6]), 0.0f);
            rv[7] = (bf16_t)fmaxf(fmaf(f1.w, A[7], B[7]), 0.0f);
        }
        *(v8bf*)(smem + pos * strideB + ch * 2) = rv;
    }
    __syncthreads();

    // ---- per-lane WMMA fragment geometry (ISA 16-bit A / B layouts) ----
    const int lane = tid & 31;
    const int wv   = tid >> 5;                 // wave id (0..1)
    const int m    = lane & 15;
    const int mt0  = wv << 1;                  // this wave's first M tile
    const int lh0  = (mt0 << 1) + (m >> 3);    // local row for M-tile mt0
    const int lw   = m & 7;
    const int klo  = (lane < 16) ? 0 : 8;      // A: lanes>=16 hold K {8-15,24-31}
    const int nlo  = (lane < 16) ? 0 : 8;      // B: lanes>=16 hold K {8-15,24-31}
    const int ncol = lane & 15;

    v8f acc00 = {}, acc01 = {}, acc10 = {}, acc11 = {};

    const int rowB = 10 * strideB;             // bytes per patch row
    for (int t = 0; t < 9; ++t) {
        const int r = t / 3, s = t - 3 * r;
        const char*   arow0 = smem + ((lh0 + r) * 10 + (lw + s)) * strideB;
        const char*   arow1 = arow0 + 2 * rowB;            // M-tile mt0+1 is 2 rows down
        const bf16_t* wt    = wpk + (size_t)t * nq * 1024;
        if (t < 8)                                          // global_prefetch next tap's B
            __builtin_prefetch(wpk + (size_t)(t + 1) * nq * 1024 + (lane << 5), 0, 1);
        for (int q = 0; q < nq; ++q) {
            const char* ap0 = arow0 + (((q << 5) + klo) << 1);
            const char* ap1 = arow1 + (((q << 5) + klo) << 1);
            v8bf a0l = *(const v8bf*)(ap0);
            v8bf a0h = *(const v8bf*)(ap0 + 32);            // +16 channels
            v8bf a1l = *(const v8bf*)(ap1);
            v8bf a1h = *(const v8bf*)(ap1 + 32);
            v16bf a0 = __builtin_shufflevector(a0l, a0h, 0, 1, 2, 3, 4, 5, 6, 7,
                                               8, 9, 10, 11, 12, 13, 14, 15);
            v16bf a1 = __builtin_shufflevector(a1l, a1h, 0, 1, 2, 3, 4, 5, 6, 7,
                                               8, 9, 10, 11, 12, 13, 14, 15);

            const bf16_t* wq  = wt + (q << 10);
            const bf16_t* b0p = wq + ncol * 32 + nlo;          // N-tile 0
            const bf16_t* b1p = wq + (16 + ncol) * 32 + nlo;   // N-tile 1
            v8bf b0l = *(const v8bf*)(b0p);
            v8bf b0h = *(const v8bf*)(b0p + 16);
            v8bf b1l = *(const v8bf*)(b1p);
            v8bf b1h = *(const v8bf*)(b1p + 16);
            v16bf b0 = __builtin_shufflevector(b0l, b0h, 0, 1, 2, 3, 4, 5, 6, 7,
                                               8, 9, 10, 11, 12, 13, 14, 15);
            v16bf b1 = __builtin_shufflevector(b1l, b1h, 0, 1, 2, 3, 4, 5, 6, 7,
                                               8, 9, 10, 11, 12, 13, 14, 15);

            acc00 = __builtin_amdgcn_wmma_f32_16x16x32_bf16(false, a0, false, b0,
                                                            (short)0, acc00, false, false);
            acc01 = __builtin_amdgcn_wmma_f32_16x16x32_bf16(false, a0, false, b1,
                                                            (short)0, acc01, false, false);
            acc10 = __builtin_amdgcn_wmma_f32_16x16x32_bf16(false, a1, false, b0,
                                                            (short)0, acc10, false, false);
            acc11 = __builtin_amdgcn_wmma_f32_16x16x32_bf16(false, a1, false, b1,
                                                            (short)0, acc11, false, false);
        }
    }

    // ---- scatter D (C/D layout: VGPR r -> M=r (+8 for lanes>=16), lane -> N) ----
    const float bia0  = bias[ncol];
    const float bia1  = bias[16 + ncol];
    const int   mbase = (lane >= 16) ? 8 : 0;
#pragma unroll
    for (int g = 0; g < 2; ++g) {
        const v8f aN0 = g ? acc10 : acc00;
        const v8f aN1 = g ? acc11 : acc01;
        const int mt  = mt0 + g;
#pragma unroll
        for (int rr = 0; rr < 8; ++rr) {
            int p  = (mt << 4) + rr + mbase;
            int gh = h0 + (p >> 3);
            int gw = w0 + (p & 7);
            float* po = outb + (((long)n * HH + gh) * WW + gw) * C_OUT + c;
            po[ncol]      = aN0[rr] + bia0;
            po[16 + ncol] = aN1[rr] + bia1;
        }
    }
}

// ---------------------------------------------------------------------------
extern "C" void kernel_launch(void* const* d_in, const int* in_sizes, int n_in,
                              void* d_out, int out_size, void* d_ws, size_t ws_size,
                              hipStream_t stream) {
    (void)in_sizes; (void)n_in; (void)out_size; (void)ws_size;
    const float* x = (const float*)d_in[0];
    const float* gam[LAYERS]; const float* bet[LAYERS]; const float* mea[LAYERS];
    const float* var[LAYERS]; const float* ker[LAYERS]; const float* bia[LAYERS];
    for (int i = 0; i < LAYERS; ++i) {
        gam[i] = (const float*)d_in[1 + i];
        bet[i] = (const float*)d_in[7 + i];
        mea[i] = (const float*)d_in[13 + i];
        var[i] = (const float*)d_in[19 + i];
        ker[i] = (const float*)d_in[25 + i];
        bia[i] = (const float*)d_in[31 + i];
    }
    float* out = (float*)d_out;
    char*  ws  = (char*)d_ws;

    // workspace layout: per-layer BN a/b (fp32), then packed bf16 weights
    float*  bnAp[LAYERS]; float* bnBp[LAYERS]; bf16_t* wp[LAYERS];
    size_t off = 0;
    for (int i = 0; i < LAYERS; ++i) {
        int c = C_IN + GROW * i;
        bnAp[i] = (float*)(ws + off); off += (size_t)c * 4;
        bnBp[i] = (float*)(ws + off); off += (size_t)c * 4;
    }
    off = (off + 255) & ~(size_t)255;
    for (int i = 0; i < LAYERS; ++i) {
        int c = C_IN + GROW * i;
        wp[i] = (bf16_t*)(ws + off); off += (size_t)9 * c * 32 * 2;
    }

    // prep (recomputed every call; deterministic, graph-capture safe)
    for (int i = 0; i < LAYERS; ++i) {
        int c = C_IN + GROW * i;
        bn_prep_kernel<<<(c + 127) / 128, 128, 0, stream>>>(gam[i], bet[i], mea[i],
                                                            var[i], bnAp[i], bnBp[i], c);
        int total = 9 * (c >> 5) * 1024;
        pack_w_kernel<<<(total + 255) / 256, 256, 0, stream>>>(ker[i], wp[i], c);
    }
    copy_x_kernel<<<2048, 256, 0, stream>>>((const float4*)x, out);

    for (int i = 0; i < LAYERS; ++i) {
        int c = C_IN + GROW * i;
        size_t lds = 100 * (size_t)(c * 2 + 16);
        dense_layer_kernel<<<dim3(7, 7, NIMG), 64, lds, stream>>>(
            out, out, wp[i], bnAp[i], bnBp[i], bia[i], c);
    }
}
